// xLSTMCell_88639535055293
// MI455X (gfx1250) — matrix operand order repaired and checked by reading, not compiled, not run on hardware
//
#include <hip/hip_runtime.h>
#include <hip/hip_bf16.h>
#include <stdint.h>

// ---------------------------------------------------------------------------
// LSTM cell: gates = [x|h_prev] @ W'^T + b'   (W' gate-interleaved: row 4h+g)
//            c_t = sig(gf)*c_prev + sig(gi)*tanh(gc);  h_t = sig(go)*tanh(c_t)
// GEMM M=8192, N=4096, K=2048, bf16 WMMA (f32 acc), async-LDS double buffer.
// ---------------------------------------------------------------------------

#define BATCH   8192
#define IN_SZ   1024
#define HID     1024
#define KDIM    2048            // IN_SZ + HID
#define N4      4096            // 4 * HID, gate-interleaved: n' = 4*h + g

typedef __attribute__((ext_vector_type(16))) __bf16 v16bf;
typedef __attribute__((ext_vector_type(8)))  float  v8f;

union FragAB {
    uint4 u[2];
    v16bf v;
};

// ---- CDNA5 async global->LDS path (guarded; falls back to sync staging) ---
#if defined(__has_builtin)
#  if __has_builtin(__builtin_amdgcn_global_load_async_to_lds_b128)
#    define HAVE_ASYNC_LDS 1
#  endif
#endif
#ifndef HAVE_ASYNC_LDS
#  define HAVE_ASYNC_LDS 0
#endif

#if HAVE_ASYNC_LDS
typedef int vec4i __attribute__((ext_vector_type(4)));
typedef __attribute__((address_space(1))) vec4i gvec4i;   // global int4
typedef __attribute__((address_space(3))) vec4i lvec4i;   // LDS int4
__device__ __forceinline__ void async_copy16(const void* g, const void* l) {
    // integer round-trips always compile; LDS generic addr low 32 bits = LDS offset
    gvec4i* gp = (gvec4i*)(uintptr_t)g;
    lvec4i* lp = (lvec4i*)(unsigned int)(uintptr_t)l;
    __builtin_amdgcn_global_load_async_to_lds_b128(gp, lp, 0, 0);
}
#  if __has_builtin(__builtin_amdgcn_s_wait_asynccnt)
#    define ASYNC_WAIT(n) __builtin_amdgcn_s_wait_asynccnt(n)
#  else
#    define ASYNC_WAIT(n) asm volatile("s_wait_asynccnt %0" :: "i"(n) : "memory")
#  endif
#endif

__device__ __forceinline__ unsigned short f2bf_rne(float f) {
    unsigned int u = __float_as_uint(f);
    u += 0x7FFFu + ((u >> 16) & 1u);   // round-to-nearest-even
    return (unsigned short)(u >> 16);
}

__device__ __forceinline__ float sigmoid_f(float x) {
    return __builtin_amdgcn_rcpf(1.0f + __expf(-x));
}

__device__ __forceinline__ float tanh_f(float x) {
    x = fminf(15.0f, fmaxf(-15.0f, x));
    float e = __expf(2.0f * x);
    return (e - 1.0f) * __builtin_amdgcn_rcpf(e + 1.0f);
}

// --------------------- prep: combined = bf16([x | h_prev]) -----------------
__global__ __launch_bounds__(256) void prep_combined_kernel(
    const float* __restrict__ x, const float* __restrict__ h_prev,
    unsigned short* __restrict__ combined)
{
    size_t idx4 = ((size_t)blockIdx.x * blockDim.x + threadIdx.x) * 4;
    if (idx4 >= (size_t)BATCH * KDIM) return;
    size_t b = idx4 >> 11;           // / KDIM
    size_t k = idx4 & (KDIM - 1);
    const float* src = (k < IN_SZ) ? (x + b * IN_SZ + k)
                                   : (h_prev + b * HID + (k - IN_SZ));
    float4 f = *reinterpret_cast<const float4*>(src);
    ushort4 o;
    o.x = f2bf_rne(f.x); o.y = f2bf_rne(f.y);
    o.z = f2bf_rne(f.z); o.w = f2bf_rne(f.w);
    *reinterpret_cast<ushort4*>(combined + idx4) = o;
}

// ------------- prep: Wp[4h+g][k] = bf16(W_g[h][k]) (gate-interleaved) ------
__global__ __launch_bounds__(256) void prep_weights_kernel(
    const float* __restrict__ Wi, const float* __restrict__ Wf,
    const float* __restrict__ Wo, const float* __restrict__ Wc,
    unsigned short* __restrict__ Wp)
{
    size_t idx4 = ((size_t)blockIdx.x * blockDim.x + threadIdx.x) * 4;
    if (idx4 >= (size_t)N4 * KDIM) return;
    size_t np = idx4 >> 11;          // n' = 4h+g
    size_t k  = idx4 & (KDIM - 1);
    size_t h  = np >> 2;
    int    g  = (int)(np & 3);
    const float* W = (g == 0) ? Wi : (g == 1) ? Wf : (g == 2) ? Wo : Wc;
    float4 f = *reinterpret_cast<const float4*>(W + h * KDIM + k);
    ushort4 o;
    o.x = f2bf_rne(f.x); o.y = f2bf_rne(f.y);
    o.z = f2bf_rne(f.z); o.w = f2bf_rne(f.w);
    *reinterpret_cast<ushort4*>(Wp + idx4) = o;
}

__global__ __launch_bounds__(256) void prep_bias_kernel(
    const float* __restrict__ bi, const float* __restrict__ bf,
    const float* __restrict__ bo, const float* __restrict__ bc,
    float* __restrict__ bp)
{
    int np = blockIdx.x * blockDim.x + threadIdx.x;
    if (np >= N4) return;
    int h = np >> 2, g = np & 3;
    const float* b = (g == 0) ? bi : (g == 1) ? bf : (g == 2) ? bo : bc;
    bp[np] = b[h];
}

// --------------------------- main fused GEMM kernel ------------------------
// Block tile 128(M) x 256(N'), BK=32, double-buffered LDS.
// 8 waves (2 over M x 4 over N), wave tile 64x64 -> 4x4 WMMA tiles.
#define BM 128
#define BN 256
#define BK 32
#define LDS_STRIDE 40            // 32 + 8 pad: 80B row stride, 16B aligned
#define NKT (KDIM / BK)          // 64 k-steps

__global__ __launch_bounds__(256) void lstm_gemm_wmma_kernel(
    const unsigned short* __restrict__ combined,  // [BATCH][KDIM] bf16
    const unsigned short* __restrict__ Wp,        // [N4][KDIM]    bf16
    const float* __restrict__ biasp,              // [N4]
    const float* __restrict__ c_prev,             // [BATCH][HID]
    float* __restrict__ out_h,                    // [BATCH][HID]
    float* __restrict__ out_c)                    // [BATCH][HID]
{
    __shared__ unsigned short As[2][BM][LDS_STRIDE];   // 20 KB
    __shared__ unsigned short Bs[2][BN][LDS_STRIDE];   // 40 KB

    const int tid    = threadIdx.x;
    const int lane   = tid & 31;
    const int laneLo = lane & 15;
    const int half   = lane >> 4;
    const int waveId = tid >> 5;
    const int waveM  = waveId >> 2;     // 0..1  (64 rows each)
    const int waveN  = waveId & 3;      // 0..3  (64 cols each)

    const int m0 = blockIdx.y * BM;     // batch tile
    const int n0 = blockIdx.x * BN;     // gate-interleaved n' tile

    v8f acc[4][4];
    #pragma unroll
    for (int mi = 0; mi < 4; ++mi)
        #pragma unroll
        for (int ni = 0; ni < 4; ++ni)
            acc[mi][ni] = (v8f){0.f, 0.f, 0.f, 0.f, 0.f, 0.f, 0.f, 0.f};

    // ---- stage issue: A tile 128x32 (512 16B chunks, 2/thread)
    //                   B tile 256x32 (1024 16B chunks, 4/thread)
    auto issue_stage = [&](int kt, int buf) {
        const int k0 = kt * BK;
#if HAVE_ASYNC_LDS
        #pragma unroll
        for (int i = 0; i < 2; ++i) {
            int ci = tid + 256 * i, row = ci >> 2, c16 = ci & 3;
            async_copy16(combined + (size_t)(m0 + row) * KDIM + k0 + c16 * 8,
                         &As[buf][row][c16 * 8]);
        }
        #pragma unroll
        for (int i = 0; i < 4; ++i) {
            int ci = tid + 256 * i, row = ci >> 2, c16 = ci & 3;
            async_copy16(Wp + (size_t)(n0 + row) * KDIM + k0 + c16 * 8,
                         &Bs[buf][row][c16 * 8]);
        }
#else
        uint4 ra[2], rb[4];
        #pragma unroll
        for (int i = 0; i < 2; ++i) {
            int ci = tid + 256 * i, row = ci >> 2, c16 = ci & 3;
            ra[i] = *reinterpret_cast<const uint4*>(
                combined + (size_t)(m0 + row) * KDIM + k0 + c16 * 8);
        }
        #pragma unroll
        for (int i = 0; i < 4; ++i) {
            int ci = tid + 256 * i, row = ci >> 2, c16 = ci & 3;
            rb[i] = *reinterpret_cast<const uint4*>(
                Wp + (size_t)(n0 + row) * KDIM + k0 + c16 * 8);
        }
        #pragma unroll
        for (int i = 0; i < 2; ++i) {
            int ci = tid + 256 * i, row = ci >> 2, c16 = ci & 3;
            *reinterpret_cast<uint4*>(&As[buf][row][c16 * 8]) = ra[i];
        }
        #pragma unroll
        for (int i = 0; i < 4; ++i) {
            int ci = tid + 256 * i, row = ci >> 2, c16 = ci & 3;
            *reinterpret_cast<uint4*>(&Bs[buf][row][c16 * 8]) = rb[i];
        }
#endif
    };

    issue_stage(0, 0);

    #pragma unroll 1
    for (int kt = 0; kt < NKT; ++kt) {
        const int buf = kt & 1;
        if (kt + 1 < NKT) {
            issue_stage(kt + 1, buf ^ 1);   // WAR safe: trailing barrier of kt-1
#if HAVE_ASYNC_LDS
            ASYNC_WAIT(6);                  // in-order: stage kt complete
#endif
        } else {
#if HAVE_ASYNC_LDS
            ASYNC_WAIT(0);
#endif
        }
        __syncthreads();                    // all waves' stage kt visible

        // B fragments: lane holds col n = laneLo, K = half*8..+7, 16+half*8..+7
        FragAB bfrag[4];
        #pragma unroll
        for (int ni = 0; ni < 4; ++ni) {
            int col = waveN * 64 + ni * 16 + laneLo;
            bfrag[ni].u[0] = *reinterpret_cast<const uint4*>(&Bs[buf][col][half * 8]);
            bfrag[ni].u[1] = *reinterpret_cast<const uint4*>(&Bs[buf][col][16 + half * 8]);
        }
        // A fragments: lane row = laneLo; V0-3: K=half*8..+7, V4-7: K=16+half*8..+7
        #pragma unroll
        for (int mi = 0; mi < 4; ++mi) {
            int row = waveM * 64 + mi * 16 + laneLo;
            FragAB afrag;
            afrag.u[0] = *reinterpret_cast<const uint4*>(&As[buf][row][half * 8]);
            afrag.u[1] = *reinterpret_cast<const uint4*>(&As[buf][row][16 + half * 8]);
            #pragma unroll
            for (int ni = 0; ni < 4; ++ni) {
                acc[mi][ni] = __builtin_amdgcn_wmma_f32_16x16x32_bf16(
                    false, afrag.v, false, bfrag[ni].v,
                    (short)0, acc[mi][ni], false, false);
            }
        }
        __syncthreads();                    // reads of buf done before overwrite
    }

    // ---------------- fused epilogue --------------------------------------
    // C layout: VGPR j, lanes 0-15: (M=j, N=lane); lanes 16-31: (M=j+8, N=lane-16)
    // n' = 4h+g : gates of one (m,h) sit in 4 adjacent lanes -> wave shuffles.
    const int base = lane & ~3;
    const int g    = lane & 3;
    #pragma unroll
    for (int mi = 0; mi < 4; ++mi) {
        #pragma unroll
        for (int ni = 0; ni < 4; ++ni) {
            int nprime = n0 + waveN * 64 + ni * 16 + laneLo;
            float bv   = biasp[nprime];
            int   h    = nprime >> 2;
            #pragma unroll
            for (int j = 0; j < 8; ++j) {
                int   m  = m0 + waveM * 64 + mi * 16 + j + 8 * half;
                float gv = acc[mi][ni][j] + bv;
                float gi = __shfl(gv, base + 0, 32);
                float gf = __shfl(gv, base + 1, 32);
                float go = __shfl(gv, base + 2, 32);
                float gc = __shfl(gv, base + 3, 32);
                float it = sigmoid_f(gi);
                float ft = sigmoid_f(gf);
                float ot = sigmoid_f(go);
                float cd = tanh_f(gc);
                float cp = c_prev[(size_t)m * HID + h];
                float cn = ft * cp + it * cd;
                float ht = ot * tanh_f(cn);
                if (g == 0) out_h[(size_t)m * HID + h] = ht;
                else if (g == 1) out_c[(size_t)m * HID + h] = cn;
            }
        }
    }
}

// ---------------------------------------------------------------------------
extern "C" void kernel_launch(void* const* d_in, const int* in_sizes, int n_in,
                              void* d_out, int out_size, void* d_ws, size_t ws_size,
                              hipStream_t stream) {
    (void)in_sizes; (void)n_in; (void)out_size; (void)ws_size;
    const float* x      = (const float*)d_in[0];
    const float* h_prev = (const float*)d_in[1];
    const float* c_prev = (const float*)d_in[2];
    const float* W_i    = (const float*)d_in[3];
    const float* b_i    = (const float*)d_in[4];
    const float* W_f    = (const float*)d_in[5];
    const float* b_f    = (const float*)d_in[6];
    const float* W_o    = (const float*)d_in[7];
    const float* b_o    = (const float*)d_in[8];
    const float* W_c    = (const float*)d_in[9];
    const float* b_c    = (const float*)d_in[10];

    float* out_h = (float*)d_out;                       // [BATCH*HID]
    float* out_c = out_h + (size_t)BATCH * HID;         // [BATCH*HID]

    // workspace layout
    unsigned short* combined = (unsigned short*)d_ws;                       // 32 MB
    unsigned short* Wp       = combined + (size_t)BATCH * KDIM;             // 16 MB
    float*          biasp    = (float*)(Wp + (size_t)N4 * KDIM);            // 16 KB

    // prep
    {
        size_t n = (size_t)BATCH * KDIM / 4;
        prep_combined_kernel<<<dim3((n + 255) / 256), dim3(256), 0, stream>>>(
            x, h_prev, combined);
    }
    {
        size_t n = (size_t)N4 * KDIM / 4;
        prep_weights_kernel<<<dim3((n + 255) / 256), dim3(256), 0, stream>>>(
            W_i, W_f, W_o, W_c, Wp);
    }
    prep_bias_kernel<<<dim3(N4 / 256), dim3(256), 0, stream>>>(
        b_i, b_f, b_o, b_c, biasp);

    // fused GEMM + LSTM epilogue
    dim3 grid(N4 / BN, BATCH / BM);   // 16 x 64
    lstm_gemm_wmma_kernel<<<grid, dim3(256), 0, stream>>>(
        combined, Wp, biasp, c_prev, out_h, out_c);
}
